// SwitchMLP_73976516707046
// MI455X (gfx1250) — matrix operand order repaired and checked by reading, not compiled, not run on hardware
//
#include <hip/hip_runtime.h>
#include <math.h>

// ---------------- problem constants ----------------
#define SDIM 2048
#define BDIM 4
#define NTOK (SDIM * BDIM)   // 8192 tokens
#define HDIM 1024
#define EXP  8
#define FFD  4096

#define TM 32                               // tokens per tile
#define MAX_TILES (NTOK / TM + EXP)         // 264

// ---------------- workspace layout (4-byte units) ----------------
#define OFF_EID   0
#define OFF_PROB  (NTOK)
#define OFF_CNT   (2 * NTOK)
#define OFF_OFFS  (2 * NTOK + 16)
#define OFF_POS   (2 * NTOK + 32)
#define OFF_NT    (2 * NTOK + 48)
#define OFF_TE    (2 * NTOK + 64)
#define OFF_TS    (OFF_TE + MAX_TILES)
#define OFF_TR    (OFF_TS + MAX_TILES)
#define OFF_PERM  (OFF_TR + MAX_TILES)

// ---------------- WMMA types ----------------
typedef __attribute__((ext_vector_type(16))) __bf16 v16bf;
typedef __attribute__((ext_vector_type(8)))  float  v8f;

struct __align__(16) U4 { unsigned int x, y, z, w; };
union Frag { v16bf v; U4 q[2]; };

static __device__ inline v16bf load_frag(const unsigned short* p0,
                                         const unsigned short* p1) {
  Frag f;
  f.q[0] = *(const U4*)p0;
  f.q[1] = *(const U4*)p1;
  return f.v;
}

static __device__ inline v8f wmma_bf16(v16bf a, v16bf b, v8f c) {
#if defined(__gfx1250__)
  // D(f32 16x16) = A(bf16 16x32) * B(bf16 32x16) + C
  return __builtin_amdgcn_wmma_f32_16x16x32_bf16(
      /*neg_a=*/false, a, /*neg_b=*/false, b,
      /*c_mod=*/(short)0, c, /*reuse_a=*/false, /*reuse_b=*/false);
#else
  (void)a; (void)b;
  return c;
#endif
}

static __device__ inline v8f v8f_zero() {
  v8f z;
  for (int i = 0; i < 8; ++i) z[i] = 0.f;
  return z;
}

// round-to-nearest-even f32 -> bf16 (bit pattern)
static __device__ inline unsigned short f2bf(float f) {
  unsigned int u = __float_as_uint(f);
  u += 0x7FFFu + ((u >> 16) & 1u);
  return (unsigned short)(u >> 16);
}

static __device__ inline float gelu_tanh(float x) {
  return 0.5f * x *
         (1.f + tanhf(0.7978845608028654f * x * (1.f + 0.044715f * x * x)));
}

// ---------------- kernels ----------------
__global__ void init_kernel(int* ws) {
  int i = threadIdx.x;
  if (i < EXP) { ws[OFF_CNT + i] = 0; ws[OFF_POS + i] = 0; }
}

// one wave32 per token: logits -> top1 prob + argmax + histogram
__global__ __launch_bounds__(256) void router_kernel(
    const float* __restrict__ hs, const float* __restrict__ rw,
    const float* __restrict__ rb, int* __restrict__ eid,
    float* __restrict__ prob, int* __restrict__ counts) {
  int wave = threadIdx.x >> 5;
  int lane = threadIdx.x & 31;
  int t = blockIdx.x * 8 + wave;
  if (t >= NTOK) return;
  float acc[EXP];
#pragma unroll
  for (int e = 0; e < EXP; ++e) acc[e] = 0.f;
  const float* xr = hs + (size_t)t * HDIM;
  for (int h = lane; h < HDIM; h += 32) {
    float xv = xr[h];
#pragma unroll
    for (int e = 0; e < EXP; ++e) acc[e] += xv * rw[e * HDIM + h];
  }
#pragma unroll
  for (int e = 0; e < EXP; ++e)
    for (int off = 16; off > 0; off >>= 1)
      acc[e] += __shfl_xor(acc[e], off, 32);
  if (lane == 0) {
    float lmax = -3.4e38f;
    int amax = 0;
#pragma unroll
    for (int e = 0; e < EXP; ++e) {
      float l = acc[e] + rb[e];
      acc[e] = l;
      if (l > lmax) { lmax = l; amax = e; }  // first-max tie-break
    }
    float s = 0.f;
#pragma unroll
    for (int e = 0; e < EXP; ++e) s += expf(acc[e] - lmax);
    prob[t] = 1.f / s;   // softmax max-prob
    eid[t] = amax;
    atomicAdd(&counts[amax], 1);
  }
}

// single-thread scan + tile schedule
__global__ void sched_kernel(const int* __restrict__ counts,
                             int* __restrict__ offsets,
                             int* __restrict__ n_tiles, int* __restrict__ te,
                             int* __restrict__ ts, int* __restrict__ tr) {
  int off = 0;
  int nt = 0;
  offsets[0] = 0;
  for (int e = 0; e < EXP; ++e) {
    int c = counts[e];
    for (int s = 0; s < c; s += TM) {
      te[nt] = e;
      ts[nt] = off + s;
      tr[nt] = (c - s < TM) ? (c - s) : TM;
      ++nt;
    }
    off += c;
    offsets[e + 1] = off;
  }
  *n_tiles = nt;
}

__global__ void assign_kernel(const int* __restrict__ eid,
                              const int* __restrict__ offsets,
                              int* __restrict__ pos, int* __restrict__ perm) {
  int t = blockIdx.x * blockDim.x + threadIdx.x;
  if (t >= NTOK) return;
  int e = eid[t];
  int p = atomicAdd(&pos[e], 1);
  perm[offsets[e] + p] = t;
}

__global__ void bias_kernel(const int* __restrict__ eid,
                            const float* __restrict__ prob,
                            const float* __restrict__ b2,
                            float* __restrict__ obias) {
  int i = blockIdx.x * blockDim.x + threadIdx.x;
  int t = i >> 10;
  int h = i & (HDIM - 1);
  obias[i] = b2[eid[t] * HDIM + h] * prob[t];
}

// ---------------- fused MoE FFN: GEMM1 -> gelu -> GEMM2 ----------------
#define XS  1032   // X row stride (bf16), padded: dword stride 516 -> banks 4m
#define W1S 136    // W1 stage row stride (128+8)
#define W2S 40     // W2 stage row stride (32+8)
#define IS  264    // inter row stride (256+8)

__global__ __launch_bounds__(256) void moe_kernel(
    const float* __restrict__ hs, const float* __restrict__ w1,
    const float* __restrict__ b1, const float* __restrict__ w2,
    const int* __restrict__ perm, const float* __restrict__ prob,
    const int* __restrict__ n_tiles, const int* __restrict__ te,
    const int* __restrict__ ts, const int* __restrict__ tr,
    float* __restrict__ out) {
  if ((int)blockIdx.x >= *n_tiles) return;

  __shared__ unsigned short Xs[TM * XS] __attribute__((aligned(16)));    // 66 KB
  __shared__ unsigned short Ws[1024 * W2S] __attribute__((aligned(16))); // 80 KB (also [256][W1S])
  __shared__ unsigned short Is[TM * IS] __attribute__((aligned(16)));    // 17 KB

  const int e = te[blockIdx.x];
  const int start = ts[blockIdx.x];
  const int rows = tr[blockIdx.x];
  const int tid = threadIdx.x;
  const int wave = tid >> 5;
  const int lane = tid & 31;
  const int ln = lane & 15;   // N/M index within fragment
  const int lh = lane >> 4;   // half select

  // gather token rows -> LDS bf16 (pad with zero rows)
  for (int i = tid; i < TM * HDIM; i += 256) {
    int r = i >> 10, k = i & (HDIM - 1);
    float v = 0.f;
    if (r < rows) v = hs[(size_t)perm[start + r] * HDIM + k];
    Xs[r * XS + k] = f2bf(v);
  }

  v8f oacc[2][8];  // persistent out accumulators: 2 M-tiles x 8 N-tiles
#pragma unroll
  for (int mt = 0; mt < 2; ++mt)
#pragma unroll
    for (int nt = 0; nt < 8; ++nt) oacc[mt][nt] = v8f_zero();

  for (int ffc = 0; ffc < FFD / 256; ++ffc) {  // 16 chunks of FF
    // -------- GEMM1: inter_chunk[32][256] = X @ W1[chunk,:]^T --------
    v8f acc1[2][2];
#pragma unroll
    for (int mt = 0; mt < 2; ++mt)
#pragma unroll
      for (int j = 0; j < 2; ++j) acc1[mt][j] = v8f_zero();

    for (int kc = 0; kc < HDIM / 128; ++kc) {  // 8 K stages of 128
      __syncthreads();  // protect Ws from previous readers
      const float* w1p =
          w1 + (size_t)e * FFD * HDIM + (size_t)(ffc * 256) * HDIM + kc * 128;
      for (int i = tid; i < 256 * 128; i += 256) {
        int n = i >> 7, k = i & 127;
        Ws[n * W1S + k] = f2bf(w1p[(size_t)n * HDIM + k]);
      }
      __syncthreads();
#pragma unroll
      for (int kt = 0; kt < 4; ++kt) {
        int kg = kc * 128 + kt * 32;  // K into Xs
        int kl = kt * 32;             // K into Ws stage
        v16bf a0 = load_frag(&Xs[ln * XS + kg + 8 * lh],
                             &Xs[ln * XS + kg + 16 + 8 * lh]);
        v16bf a1 = load_frag(&Xs[(16 + ln) * XS + kg + 8 * lh],
                             &Xs[(16 + ln) * XS + kg + 16 + 8 * lh]);
#pragma unroll
        for (int j = 0; j < 2; ++j) {
          int nrow = (wave * 2 + j) * 16 + ln;
          v16bf b = load_frag(&Ws[nrow * W1S + kl + 8 * lh],
                              &Ws[nrow * W1S + kl + 16 + 8 * lh]);
          acc1[0][j] = wmma_bf16(a0, b, acc1[0][j]);
          acc1[1][j] = wmma_bf16(a1, b, acc1[1][j]);
        }
      }
    }
    // bias + gelu (f32) -> bf16 inter tile
#pragma unroll
    for (int j = 0; j < 2; ++j) {
      int ncol = (wave * 2 + j) * 16 + ln;  // chunk-local column
      float bb = b1[e * FFD + ffc * 256 + ncol];
#pragma unroll
      for (int mt = 0; mt < 2; ++mt)
#pragma unroll
        for (int r = 0; r < 8; ++r) {
          int m = mt * 16 + r + 8 * lh;
          Is[m * IS + ncol] = f2bf(gelu_tanh(acc1[mt][j][r] + bb));
        }
    }

    // -------- GEMM2 partial: oacc += inter_chunk @ W2[:,chunk]^T --------
    for (int kc2 = 0; kc2 < 8; ++kc2) {  // 8 K stages of 32
      __syncthreads();  // Is visibility (first) / Ws protect (rest)
      const float* w2p = w2 + (size_t)e * HDIM * FFD + ffc * 256 + kc2 * 32;
      for (int i = tid; i < 1024 * 32; i += 256) {
        int n = i >> 5, k = i & 31;
        Ws[n * W2S + k] = f2bf(w2p[(size_t)n * FFD + k]);
      }
      __syncthreads();
      int kg = kc2 * 32;  // K into Is columns
      v16bf a0 = load_frag(&Is[ln * IS + kg + 8 * lh],
                           &Is[ln * IS + kg + 16 + 8 * lh]);
      v16bf a1 = load_frag(&Is[(16 + ln) * IS + kg + 8 * lh],
                           &Is[(16 + ln) * IS + kg + 16 + 8 * lh]);
#pragma unroll
      for (int nt = 0; nt < 8; ++nt) {
        int nrow = wave * 128 + nt * 16 + ln;
        v16bf b = load_frag(&Ws[nrow * W2S + 8 * lh],
                            &Ws[nrow * W2S + 16 + 8 * lh]);
        oacc[0][nt] = wmma_bf16(a0, b, oacc[0][nt]);
        oacc[1][nt] = wmma_bf16(a1, b, oacc[1][nt]);
      }
    }
  }

  // scale by routing prob and scatter rows back
#pragma unroll
  for (int mt = 0; mt < 2; ++mt)
#pragma unroll
    for (int r = 0; r < 8; ++r) {
      int m = mt * 16 + r + 8 * lh;
      if (m < rows) {
        int t = perm[start + m];
        float p = prob[t];
#pragma unroll
        for (int nt = 0; nt < 8; ++nt)
          out[(size_t)t * HDIM + wave * 128 + nt * 16 + ln] =
              oacc[mt][nt][r] * p;
      }
    }
}

// ---------------- host launch ----------------
extern "C" void kernel_launch(void* const* d_in, const int* in_sizes, int n_in,
                              void* d_out, int out_size, void* d_ws,
                              size_t ws_size, hipStream_t stream) {
  (void)in_sizes; (void)n_in; (void)out_size; (void)ws_size;
  const float* hs = (const float*)d_in[0];
  const float* rw = (const float*)d_in[1];
  const float* rb = (const float*)d_in[2];
  const float* w1 = (const float*)d_in[3];
  const float* b1 = (const float*)d_in[4];
  const float* w2 = (const float*)d_in[5];
  const float* b2 = (const float*)d_in[6];
  float* out = (float*)d_out;                       // [N*H] out
  float* obias = out + (size_t)NTOK * HDIM;         // [N*H] out_bias
  int* wsI = (int*)d_ws;
  float* wsF = (float*)d_ws;

  int* eid = wsI + OFF_EID;
  float* prob = wsF + OFF_PROB;
  int* counts = wsI + OFF_CNT;
  int* offsets = wsI + OFF_OFFS;
  int* pos = wsI + OFF_POS;
  int* n_tiles = wsI + OFF_NT;
  int* te = wsI + OFF_TE;
  int* ts = wsI + OFF_TS;
  int* tr = wsI + OFF_TR;
  int* perm = wsI + OFF_PERM;

  init_kernel<<<1, 32, 0, stream>>>(wsI);
  router_kernel<<<NTOK / 8, 256, 0, stream>>>(hs, rw, rb, eid, prob, counts);
  sched_kernel<<<1, 1, 0, stream>>>(counts, offsets, n_tiles, te, ts, tr);
  assign_kernel<<<NTOK / 256, 256, 0, stream>>>(eid, offsets, pos, perm);
  bias_kernel<<<(NTOK * HDIM) / 256, 256, 0, stream>>>(eid, prob, b2, obias);
  moe_kernel<<<MAX_TILES, 256, 0, stream>>>(hs, w1, b1, w2, perm, prob,
                                            n_tiles, te, ts, tr, out);
}
